// LSTMModel_5712306503767
// MI455X (gfx1250) — matrix operand order repaired and checked by reading, not compiled, not run on hardware
//
#include <hip/hip_runtime.h>
#include <hip/hip_bf16.h>

// ============================================================================
// Fused 2-layer LSTM (H=50 pad->64) + FC head for MI455X (gfx1250, wave32).
//
//  * one wave = 16 batch rows, runs all T=512 steps of both layers fused
//  * all GEMMs are v_wmma_f32_16x16x32_f16 (f16 in, f32 acc)
//  * gate layout N=256 (i@0,f@64,g@128,o@192) -> all 4 gates of a unit are
//    lane-local in the C layout -> no cross-lane traffic for the nonlinearity
//  * layer0 A = [h0 | x_t in K slots 50,51] (K=64); layer1 A = [h1 | h0] (K=128)
//  * weights reordered once into LDS as f16, column-major K-contiguous with
//    padded strides (72 / 136 halves) -> B operands are two ds_load_b128
//  * h C->A transpose via wave-private LDS buffer; same-wave LDS is in-order,
//    so the T loop has NO barriers
//  * cell state c0/c1 lives in VGPRs across the whole sequence
//  * an asm memory clobber per timestep stops LICM from hoisting the 768
//    VGPRs worth of loop-invariant B-operands out of the loop (which caused
//    scratch spill/reload inside the recurrence in the previous revision);
//    weights are intended to STREAM from LDS every step.
// ============================================================================

#define LSTM_H   50
#define HP       64
#define SEQ_T    512
#define DIN      2
#define NCOL     256               // padded gate columns (16 n-tiles)
#define W0_STR   72                // halves per column of W0 (64 K + pad)
#define W1_STR   136               // halves per column of W1 (128 K + pad)
#define A_STR    136               // halves per row of A buffer (128 K + pad)
#define A_ELEMS  (16 * A_STR)
#define W0_ELEMS (NCOL * W0_STR)
#define W1_ELEMS (NCOL * W1_STR)
#define NWAVES   2
#define BLOCK    (NWAVES * 32)

typedef _Float16 h8   __attribute__((ext_vector_type(8)));
typedef _Float16 v16h __attribute__((ext_vector_type(16)));
typedef float    v8f  __attribute__((ext_vector_type(8)));

// Assemble a 16-half WMMA operand from two 16-byte LDS chunks (ds_load_b128).
__device__ __forceinline__ v16h ld_op(const _Float16* p0, const _Float16* p1) {
  h8 lo, hi;
  __builtin_memcpy(&lo, p0, sizeof(h8));
  __builtin_memcpy(&hi, p1, sizeof(h8));
  return __builtin_shufflevector(lo, hi, 0, 1, 2, 3, 4, 5, 6, 7,
                                         8, 9, 10, 11, 12, 13, 14, 15);
}

__device__ __forceinline__ float fexp(float x) {
  return __builtin_amdgcn_exp2f(x * 1.44269504088896340736f);   // v_exp_f32
}
__device__ __forceinline__ float sigm(float x) {
  return __builtin_amdgcn_rcpf(1.0f + fexp(-x));                // v_rcp_f32
}
__device__ __forceinline__ float tanh_f(float x) {
  return 2.0f * __builtin_amdgcn_rcpf(1.0f + fexp(-2.0f * x)) - 1.0f;
}

__global__ void __launch_bounds__(BLOCK, 1)
lstm2_wmma_kernel(const float* __restrict__ x,
                  const float* __restrict__ Wih0, const float* __restrict__ Whh0,
                  const float* __restrict__ b0,
                  const float* __restrict__ Wih1, const float* __restrict__ Whh1,
                  const float* __restrict__ b1,
                  const float* __restrict__ Wfc,  const float* __restrict__ bfc,
                  float* __restrict__ out)
{
  __shared__ __align__(16) _Float16 sW0[W0_ELEMS];          // 36.0 KB
  __shared__ __align__(16) _Float16 sW1[W1_ELEMS];          // 68.0 KB
  __shared__ __align__(16) _Float16 sA[NWAVES * A_ELEMS];   //  8.5 KB
  __shared__ float sB0[NCOL];                               //  1 KB
  __shared__ float sB1[NCOL];                               //  1 KB

  const int tid = threadIdx.x;

  // ---- Stage + reorder weights into LDS (once; reused 512 timesteps) ------
  // Column c: gate g=c>>6, unit j=c&63 (zero-pad j>=50). K-contiguous per col.
  for (int i = tid; i < W0_ELEMS; i += BLOCK) {
    int c = i / W0_STR, k = i - c * W0_STR;
    int g = c >> 6, j = c & 63;
    float w = 0.0f;
    if (j < LSTM_H) {
      int r = g * LSTM_H + j;                       // row of original [200 x *]
      if (k < LSTM_H)            w = Whh0[r * LSTM_H + k];        // K 0..49
      else if (k < LSTM_H + DIN) w = Wih0[r * DIN + (k - LSTM_H)];// K 50..51 = x
    }
    sW0[i] = (_Float16)w;
  }
  for (int i = tid; i < W1_ELEMS; i += BLOCK) {
    int c = i / W1_STR, k = i - c * W1_STR;
    int g = c >> 6, j = c & 63;
    float w = 0.0f;
    if (j < LSTM_H) {
      int r = g * LSTM_H + j;
      if (k < LSTM_H)                      w = Whh1[r * LSTM_H + k];        // h1
      else if (k >= HP && k < HP + LSTM_H) w = Wih1[r * LSTM_H + (k - HP)]; // h0
    }
    sW1[i] = (_Float16)w;
  }
  for (int i = tid; i < NWAVES * A_ELEMS; i += BLOCK) sA[i] = (_Float16)0.0f;
  for (int i = tid; i < NCOL; i += BLOCK) {
    int g = i >> 6, j = i & 63;
    sB0[i] = (j < LSTM_H) ? b0[g * LSTM_H + j] : 0.0f;
    sB1[i] = (j < LSTM_H) ? b1[g * LSTM_H + j] : 0.0f;
  }
  __syncthreads();

  // ---- per-lane geometry ---------------------------------------------------
  const int lane = tid & 31;
  const int wv   = tid >> 5;
  const int lm   = lane & 15;            // C/B layout: N-within-tile, A: row M
  const int lh   = lane >> 4;            // lane half
  const int aklo = lh * 8;               // A operand K sub-offset
  const int bklo = lh * 8;               // B operand K sub-offset
  _Float16* A    = sA + wv * A_ELEMS;    // wave-private [16 rows][A_STR]
  _Float16* Arow = A + lm * A_STR;       // this lane's A-operand row

  const int rowBase = blockIdx.x * (16 * NWAVES) + wv * 16;
  const float* xp = x + (size_t)(rowBase + lm) * SEQ_T * DIN;

  v8f c0[4], c1[4];                      // cell state, lane-local, f32
  const v8f vzero = {0.f, 0.f, 0.f, 0.f, 0.f, 0.f, 0.f, 0.f};
#pragma unroll
  for (int n = 0; n < 4; ++n) { c0[n] = vzero; c1[n] = vzero; }

  // ---- time loop (no barriers: A buffer is wave-private, LDS in-order) ----
  for (int t = 0; t < SEQ_T; ++t) {
    // Opaque memory clobber: keeps every weight ds_load inside the loop
    // (prevents LICM from hoisting 96 B-operands -> scratch spills).
    asm volatile("" ::: "memory");

    // stage x_t into K slots 50,51 of the h0 half (lanes 0..15, one per row)
    if (lane < 16) {
      float xv[2];
      __builtin_memcpy(xv, xp + t * DIN, 8);
      _Float16 xh[2] = {(_Float16)xv[0], (_Float16)xv[1]};
      __builtin_memcpy(A + lane * A_STR + (HP + LSTM_H), xh, 4); // ds_store_b32
    }

    v8f acc[16];

    // ====== layer 0: gates0 = [h0_{t-1} | x_t] @ W0   (K=64, 32 WMMAs) =====
#pragma unroll
    for (int n = 0; n < 16; ++n) acc[n] = vzero;
#pragma unroll
    for (int kc = 0; kc < 2; ++kc) {
      const int ka = HP + kc * 32 + aklo;
      v16h av = ld_op(Arow + ka, Arow + ka + 16);
      const int kb = kc * 32 + bklo;
#pragma unroll
      for (int n = 0; n < 16; ++n) {
        const _Float16* bp = sW0 + (n * 16 + lm) * W0_STR + kb;
        v16h bv = ld_op(bp, bp + 16);
        acc[n] = __builtin_amdgcn_wmma_f32_16x16x32_f16(
            false, av, false, bv, (short)0, acc[n], false, false);
      }
    }
    // gate nonlinearity + c0 update + h0 -> LDS (K base 64)
#pragma unroll
    for (int n = 0; n < 4; ++n) {
      const float bi = sB0[n * 16 + lm];
      const float bf = sB0[64 + n * 16 + lm];
      const float bg = sB0[128 + n * 16 + lm];
      const float bo = sB0[192 + n * 16 + lm];
#pragma unroll
      for (int v = 0; v < 8; ++v) {
        float gi = sigm(acc[n][v] + bi);
        float gf = sigm(acc[n + 4][v] + bf);
        float gg = tanh_f(acc[n + 8][v] + bg);
        float go = sigm(acc[n + 12][v] + bo);
        float cc = gf * c0[n][v] + gi * gg;
        c0[n][v] = cc;
        float hh = go * tanh_f(cc);
        A[(v + 8 * lh) * A_STR + HP + n * 16 + lm] = (_Float16)hh;
      }
    }

    // ====== layer 1: gates1 = [h1_{t-1} | h0_t] @ W1   (K=128, 64 WMMAs) ===
#pragma unroll
    for (int n = 0; n < 16; ++n) acc[n] = vzero;
#pragma unroll
    for (int kc = 0; kc < 4; ++kc) {
      const int ka = kc * 32 + aklo;
      v16h av = ld_op(Arow + ka, Arow + ka + 16);
      const int kb = kc * 32 + bklo;
#pragma unroll
      for (int n = 0; n < 16; ++n) {
        const _Float16* bp = sW1 + (n * 16 + lm) * W1_STR + kb;
        v16h bv = ld_op(bp, bp + 16);
        acc[n] = __builtin_amdgcn_wmma_f32_16x16x32_f16(
            false, av, false, bv, (short)0, acc[n], false, false);
      }
    }
#pragma unroll
    for (int n = 0; n < 4; ++n) {
      const float bi = sB1[n * 16 + lm];
      const float bf = sB1[64 + n * 16 + lm];
      const float bg = sB1[128 + n * 16 + lm];
      const float bo = sB1[192 + n * 16 + lm];
#pragma unroll
      for (int v = 0; v < 8; ++v) {
        float gi = sigm(acc[n][v] + bi);
        float gf = sigm(acc[n + 4][v] + bf);
        float gg = tanh_f(acc[n + 8][v] + bg);
        float go = sigm(acc[n + 12][v] + bo);
        float cc = gf * c1[n][v] + gi * gg;
        c1[n][v] = cc;
        float hh = go * tanh_f(cc);
        A[(v + 8 * lh) * A_STR + n * 16 + lm] = (_Float16)hh;   // h1 at K base 0
      }
    }
  }

  // ---- FC head on h1_{T-1}: out[b] = h1 . Wfc + bfc ------------------------
  float part[8] = {0.f, 0.f, 0.f, 0.f, 0.f, 0.f, 0.f, 0.f};
#pragma unroll
  for (int n = 0; n < 4; ++n) {
    const int j = n * 16 + lm;                       // hidden unit
    const float wf = (j < LSTM_H) ? Wfc[j] : 0.0f;
#pragma unroll
    for (int v = 0; v < 8; ++v) {
      float hv = (float)A[(v + 8 * lh) * A_STR + j];
      part[v] += hv * wf;
    }
  }
#pragma unroll
  for (int v = 0; v < 8; ++v) {
#pragma unroll
    for (int m = 1; m < 16; m <<= 1)
      part[v] += __shfl_xor(part[v], m, 32);         // reduce over 16 columns
  }
  if (lm == 0) {
    const float bb = bfc[0];
#pragma unroll
    for (int v = 0; v < 8; ++v)
      out[rowBase + 8 * lh + v] = part[v] + bb;      // rows v / v+8 per half
  }
}

extern "C" void kernel_launch(void* const* d_in, const int* in_sizes, int n_in,
                              void* d_out, int out_size, void* d_ws, size_t ws_size,
                              hipStream_t stream) {
  (void)n_in; (void)out_size; (void)d_ws; (void)ws_size;
  const float* x    = (const float*)d_in[0];
  const float* Wih0 = (const float*)d_in[1];
  const float* Whh0 = (const float*)d_in[2];
  const float* b0   = (const float*)d_in[3];
  const float* Wih1 = (const float*)d_in[4];
  const float* Whh1 = (const float*)d_in[5];
  const float* b1   = (const float*)d_in[6];
  const float* Wfc  = (const float*)d_in[7];
  const float* bfc  = (const float*)d_in[8];
  float* out        = (float*)d_out;

  const int B    = in_sizes[0] / (SEQ_T * DIN);     // 4096
  const int grid = B / (16 * NWAVES);               // 16 rows per wave

  lstm2_wmma_kernel<<<grid, BLOCK, 0, stream>>>(
      x, Wih0, Whh0, b0, Wih1, Whh1, b1, Wfc, bfc, out);
}
